// Gemma3Attention_3934190043257
// MI455X (gfx1250) — matrix operand order repaired and checked by reading, not compiled, not run on hardware
//
#include <hip/hip_runtime.h>
#include <stdint.h>

// ---------------- problem constants ----------------
#define S_LEN 2048
#define HID   2560
#define NHH   32
#define NKVH  8
#define HD    128
#define NQ    (NHH * HD)   // 4096
#define NKVD  (NKVH * HD)  // 1024

// ---------------- WMMA / TDM types ----------------
typedef __attribute__((ext_vector_type(16))) __bf16 v16bf;
typedef __attribute__((ext_vector_type(8)))  float  v8f;
typedef __attribute__((ext_vector_type(4)))  unsigned int v4u;
typedef __attribute__((ext_vector_type(8)))  int    v8i;
typedef __attribute__((ext_vector_type(4)))  int    v4i;

#if defined(__has_builtin)
#if __has_builtin(__builtin_amdgcn_tensor_load_to_lds) && \
    __has_builtin(__builtin_amdgcn_s_wait_tensorcnt)
#define USE_TDM 1
#endif
#endif
#ifndef USE_TDM
#define USE_TDM 0
#endif

union Frag { uint4 q[2]; v16bf v; };

__device__ __forceinline__ unsigned short f32_to_bf16(float f) {
  unsigned int u = __float_as_uint(f);
  u += 0x7fffu + ((u >> 16) & 1u);   // round-to-nearest-even
  return (unsigned short)(u >> 16);
}

__device__ __forceinline__ v8f wmma_bf16(v16bf a, v16bf b, v8f c) {
  // (neg_a, A, neg_b, B, c_mod, C, reuse_a, reuse_b)
  return __builtin_amdgcn_wmma_f32_16x16x32_bf16(false, a, false, b, (short)0, c,
                                                 false, false);
}

// ---------------- fp32 -> bf16 convert (vec4) ----------------
__global__ __launch_bounds__(256) void cvt_bf16_kernel(
    const float* __restrict__ src, unsigned short* __restrict__ dst, int n4) {
  int i = blockIdx.x * blockDim.x + threadIdx.x;
  if (i < n4) {
    float4 v = ((const float4*)src)[i];
    ushort4 o;
    o.x = f32_to_bf16(v.x); o.y = f32_to_bf16(v.y);
    o.z = f32_to_bf16(v.z); o.w = f32_to_bf16(v.w);
    ((ushort4*)dst)[i] = o;
  }
}

// ---------------- tiled WMMA GEMM: C(MxN) = A(MxK) @ B(KxN) ----------------
// A, B bf16 row-major.  Block tile 128x128, K-step 32, 8 waves (2x4),
// each wave computes 64x32 via 4x2 accumulators of 16x16.
// A tile staged via Tensor Data Mover (2D descriptor), B staged transposed.
template <bool OUT_BF16>
__global__ __launch_bounds__(256) void gemm_bf16_kernel(
    const unsigned short* __restrict__ A, const unsigned short* __restrict__ B,
    void* __restrict__ Cout, int M, int N, int K) {
  __shared__ unsigned short As[128 * 32];  // [row][k]
  __shared__ unsigned short Bt[128 * 32];  // [n][k]  (transposed stage)
  const int tid = threadIdx.x;
  const int lane = tid & 31;
  const int wave = tid >> 5;
  const int l16 = lane & 15, half = lane >> 4;
  const int wm = wave & 1, wn = wave >> 1;  // 2 x 4 wave grid
  const int bm = blockIdx.y * 128, bn = blockIdx.x * 128;

  v8f acc[4][2];
  const v8f vzero = {};
#pragma unroll
  for (int i = 0; i < 4; ++i)
#pragma unroll
    for (int j = 0; j < 2; ++j) acc[i][j] = vzero;

  const int n2 = (tid & 63) * 2;   // column pair for Bt stage
  const int kq = (tid >> 6) * 8;   // k-quarter for Bt stage

  for (int k0 = 0; k0 < K; k0 += 32) {
    // ---- stage A tile (128x32) via TDM ----
#if USE_TDM
    if (wave == 0) {
      unsigned long long ga =
          (unsigned long long)(const void*)(A + (size_t)bm * K + k0);
      unsigned int ldsa = (unsigned int)(unsigned long long)(const void*)As;
      v4u g0;
      g0[0] = 1u;                                   // count=1 (valid descriptor)
      g0[1] = ldsa;                                 // lds_addr
      g0[2] = (unsigned int)ga;                     // global_addr[31:0]
      g0[3] = (unsigned int)((ga >> 32) & 0x01FFFFFFu) | (2u << 30);  // type=2
      v8i g1;
      const unsigned uK = (unsigned)K, uM = (unsigned)M;
      g1[0] = (int)(1u << 16);                                  // data_size=2B
      g1[1] = (int)((uK & 0xFFFFu) << 16);                      // tensor_dim0 lo
      g1[2] = (int)(((uK >> 16) & 0xFFFFu) | ((uM & 0xFFFFu) << 16));
      g1[3] = (int)(((uM >> 16) & 0xFFFFu) | (32u << 16));      // tile_dim0=32
      g1[4] = (int)128u;                                        // tile_dim1=128
      g1[5] = (int)uK;                                          // dim0_stride lo
      g1[6] = 0;
      g1[7] = 0;
      v4i gz4 = {};
      v8i gz8 = {};
      __builtin_amdgcn_tensor_load_to_lds(g0, g1, gz4, gz4, gz8, 0);
    }
#else
    {
      const int arow = tid & 127, aseg = tid >> 7;
      const uint4* src =
          (const uint4*)(A + (size_t)(bm + arow) * K + k0 + aseg * 16);
      uint4* dst = (uint4*)(As + arow * 32 + aseg * 16);
      dst[0] = src[0];
      dst[1] = src[1];
    }
#endif

    // ---- stage B transposed: Bt[n][k] = B[k0+k][bn+n], packed b32 stores ----
#pragma unroll
    for (int kk = 0; kk < 8; kk += 2) {
      const int k = kq + kk;
      unsigned int a = *(const unsigned int*)(B + (size_t)(k0 + k) * N + bn + n2);
      unsigned int b =
          *(const unsigned int*)(B + (size_t)(k0 + k + 1) * N + bn + n2);
      *(unsigned int*)(Bt + n2 * 32 + k) = (a & 0xFFFFu) | (b << 16);
      *(unsigned int*)(Bt + (n2 + 1) * 32 + k) = (a >> 16) | (b & 0xFFFF0000u);
    }
    if (k0 + 32 < K) {
      __builtin_prefetch(B + (size_t)(k0 + 32 + kq) * N + bn + n2, 0, 1);
      __builtin_prefetch(A + (size_t)(bm + (tid & 127)) * K + k0 + 32, 0, 1);
    }
#if USE_TDM
    if (wave == 0) __builtin_amdgcn_s_wait_tensorcnt(0);
#endif
    __syncthreads();

    v16bf af[4];
#pragma unroll
    for (int i = 0; i < 4; ++i) {
      Frag f;
      const unsigned short* p = As + (wm * 64 + i * 16 + l16) * 32;
      f.q[0] = *(const uint4*)(p + half * 8);        // k = half*8 .. +7
      f.q[1] = *(const uint4*)(p + 16 + half * 8);   // k = 16+half*8 .. +7
      af[i] = f.v;
    }
    v16bf bfr[2];
#pragma unroll
    for (int j = 0; j < 2; ++j) {
      Frag f;
      const unsigned short* p = Bt + (wn * 32 + j * 16 + l16) * 32;
      f.q[0] = *(const uint4*)(p + half * 8);        // k = half*8 .. +7
      f.q[1] = *(const uint4*)(p + 16 + half * 8);   // k = 16+half*8 .. +7
      bfr[j] = f.v;
    }
#pragma unroll
    for (int i = 0; i < 4; ++i)
#pragma unroll
      for (int j = 0; j < 2; ++j) acc[i][j] = wmma_bf16(af[i], bfr[j], acc[i][j]);
    __syncthreads();
  }

  // store: C layout lane=N%16, element e -> M = e + 8*half
#pragma unroll
  for (int i = 0; i < 4; ++i) {
    const int mbase = bm + wm * 64 + i * 16 + 8 * half;
#pragma unroll
    for (int j = 0; j < 2; ++j) {
      const int n = bn + wn * 32 + j * 16 + l16;
#pragma unroll
      for (int e = 0; e < 8; ++e) {
        size_t idx = (size_t)(mbase + e) * N + n;
        if (OUT_BF16)
          ((unsigned short*)Cout)[idx] = f32_to_bf16(acc[i][j][e]);
        else
          ((float*)Cout)[idx] = acc[i][j][e];
      }
    }
  }
}

// ---------------- per-head RMSNorm + RoPE, fp32 -> bf16 ----------------
__global__ __launch_bounds__(256) void normrope_kernel(
    const float* __restrict__ X, const float* __restrict__ w,
    const float* __restrict__ cosT, const float* __restrict__ sinT,
    unsigned short* __restrict__ Y, int nheads) {
  const int lane = threadIdx.x & 31, wave = threadIdx.x >> 5;
  const int row = blockIdx.x * 8 + wave;
  const int s = row / nheads, h = row % nheads;
  const int stride = nheads * HD;
  const float* x = X + (size_t)s * stride + h * HD;
  const int d0 = lane * 4;

  float4 v = *(const float4*)(x + d0);
  float ss = v.x * v.x + v.y * v.y + v.z * v.z + v.w * v.w;
#pragma unroll
  for (int m = 16; m >= 1; m >>= 1) ss += __shfl_xor(ss, m, 32);
  const float inv = rsqrtf(ss * (1.0f / HD) + 1e-6f);

  float4 wv = *(const float4*)(w + d0);
  float4 pv = *(const float4*)(x + (d0 ^ 64));   // rotate_half partner
  float4 pw = *(const float4*)(w + (d0 ^ 64));
  float4 c  = *(const float4*)(cosT + (size_t)s * HD + d0);
  float4 sn = *(const float4*)(sinT + (size_t)s * HD + d0);
  const float sgn = (d0 < 64) ? -1.0f : 1.0f;

  float o0 = v.x * inv * (1.0f + wv.x) * c.x + sgn * pv.x * inv * (1.0f + pw.x) * sn.x;
  float o1 = v.y * inv * (1.0f + wv.y) * c.y + sgn * pv.y * inv * (1.0f + pw.y) * sn.y;
  float o2 = v.z * inv * (1.0f + wv.z) * c.z + sgn * pv.z * inv * (1.0f + pw.z) * sn.z;
  float o3 = v.w * inv * (1.0f + wv.w) * c.w + sgn * pv.w * inv * (1.0f + pw.w) * sn.w;

  ushort4 ov;
  ov.x = f32_to_bf16(o0); ov.y = f32_to_bf16(o1);
  ov.z = f32_to_bf16(o2); ov.w = f32_to_bf16(o3);
  *(ushort4*)(Y + (size_t)s * stride + h * HD + d0) = ov;
}

// ---------------- flash attention (causal, GQA, tanh softcap) ----------------
// grid = (NHH, S/64); block = 128 (4 waves x 16 q-rows).
__global__ __launch_bounds__(128) void attn_kernel(
    const unsigned short* __restrict__ Qb,  // S x 4096 bf16
    const unsigned short* __restrict__ Kb,  // S x 1024 bf16
    const unsigned short* __restrict__ Vb,  // S x 1024 bf16
    unsigned short* __restrict__ Ob) {      // S x 4096 bf16
  const int h = blockIdx.x;
  const int qt = blockIdx.y;
  const int kvh = h >> 2;  // GROUPS = 4
  __shared__ unsigned short Vt[HD * 32];     // [d][key]
  __shared__ unsigned short Pl[4][16 * 32];  // per-wave P staging [m][key]
  const int tid = threadIdx.x, lane = tid & 31, wave = tid >> 5;
  const int l16 = lane & 15, half = lane >> 4;
  const int qbase = qt * 64 + wave * 16;

  const float SCALING = 0.18898223650461363f;  // 28^-0.5
  const float SOFTCAP = 50.0f;

  // cache Q fragments (16 rows x 128 d) in registers
  v16bf aq[4];
  {
    const unsigned short* qp = Qb + (size_t)(qbase + l16) * NQ + h * HD;
#pragma unroll
    for (int c = 0; c < 4; ++c) {
      Frag f;
      f.q[0] = *(const uint4*)(qp + c * 32 + half * 8);
      f.q[1] = *(const uint4*)(qp + c * 32 + 16 + half * 8);
      aq[c] = f.v;
    }
  }

  v8f acc_o[8];
  const v8f vzero = {};
#pragma unroll
  for (int i = 0; i < 8; ++i) acc_o[i] = vzero;
  float rm[8], rl[8];
#pragma unroll
  for (int e = 0; e < 8; ++e) { rm[e] = -1.0e30f; rl[e] = 0.0f; }

  const int kend = qt * 64 + 64;
  for (int kb = 0; kb < kend; kb += 32) {
    __syncthreads();  // protect Vt from previous iteration's readers
    // cooperative stage of V transposed: Vt[d][key] = V[kb+key][kvh*128+d]
    // key-pairs packed into b32 stores.
    {
      const int kp2 = (tid & 15) * 2;        // keys kp2, kp2+1
      const int dseg = (tid >> 4) * 16;      // 16 d per thread
      const unsigned short* vp0 =
          Vb + (size_t)(kb + kp2) * NKVD + kvh * HD + dseg;
      const unsigned short* vp1 = vp0 + NKVD;
      union U4 { uint4 q; unsigned int w[4]; } a0, a1, b0, b1;
      a0.q = *(const uint4*)(vp0);
      b0.q = *(const uint4*)(vp0 + 8);
      a1.q = *(const uint4*)(vp1);
      b1.q = *(const uint4*)(vp1 + 8);
#pragma unroll
      for (int i = 0; i < 4; ++i) {
        unsigned int x0 = a0.w[i], x1 = a1.w[i];
        *(unsigned int*)(Vt + (dseg + 2 * i) * 32 + kp2) =
            (x0 & 0xFFFFu) | (x1 << 16);
        *(unsigned int*)(Vt + (dseg + 2 * i + 1) * 32 + kp2) =
            (x0 >> 16) | (x1 & 0xFFFF0000u);
        unsigned int y0 = b0.w[i], y1 = b1.w[i];
        *(unsigned int*)(Vt + (dseg + 8 + 2 * i) * 32 + kp2) =
            (y0 & 0xFFFFu) | (y1 << 16);
        *(unsigned int*)(Vt + (dseg + 8 + 2 * i + 1) * 32 + kp2) =
            (y0 >> 16) | (y1 & 0xFFFF0000u);
      }
    }
    __syncthreads();

    // scores: two 16x16 tiles, K^T fragments straight from global
    v8f sc[2];
    sc[0] = vzero; sc[1] = vzero;
#pragma unroll
    for (int t = 0; t < 2; ++t) {
      const unsigned short* kp = Kb + (size_t)(kb + t * 16 + l16) * NKVD + kvh * HD;
#pragma unroll
      for (int c = 0; c < 4; ++c) {
        Frag f;
        f.q[0] = *(const uint4*)(kp + c * 32 + half * 8);
        f.q[1] = *(const uint4*)(kp + c * 32 + 16 + half * 8);
        sc[t] = wmma_bf16(aq[c], f.v, sc[t]);
      }
    }

    // softcap + causal mask + online softmax (fp32)
    float pr0[8], pr1[8], osc[8];
#pragma unroll
    for (int e = 0; e < 8; ++e) {
      const int qrow = qbase + e + 8 * half;
      float s0 = tanhf(sc[0][e] * SCALING * (1.0f / SOFTCAP)) * SOFTCAP;
      float s1 = tanhf(sc[1][e] * SCALING * (1.0f / SOFTCAP)) * SOFTCAP;
      s0 += ((kb + l16) <= qrow) ? 0.0f : -1.0e9f;
      s1 += ((kb + 16 + l16) <= qrow) ? 0.0f : -1.0e9f;
      float lm = fmaxf(s0, s1);
#pragma unroll
      for (int m = 8; m >= 1; m >>= 1) lm = fmaxf(lm, __shfl_xor(lm, m, 32));
      const float nm = fmaxf(rm[e], lm);
      const float scale = __expf(rm[e] - nm);
      const float p0 = __expf(s0 - nm);
      const float p1 = __expf(s1 - nm);
      float ps = p0 + p1;
#pragma unroll
      for (int m = 8; m >= 1; m >>= 1) ps += __shfl_xor(ps, m, 32);
      rl[e] = rl[e] * scale + ps;
      rm[e] = nm;
      pr0[e] = p0; pr1[e] = p1; osc[e] = scale;
    }
#pragma unroll
    for (int sub = 0; sub < 8; ++sub)
#pragma unroll
      for (int e = 0; e < 8; ++e) acc_o[sub][e] *= osc[e];

    // re-layout P (C layout -> A layout) through per-wave LDS
    unsigned short* pb = Pl[wave];
#pragma unroll
    for (int e = 0; e < 8; ++e) {
      const int r = e + 8 * half;
      pb[r * 32 + l16]      = f32_to_bf16(pr0[e]);
      pb[r * 32 + 16 + l16] = f32_to_bf16(pr1[e]);
    }
    Frag pa;
    {
      const unsigned short* pw2 = pb + l16 * 32;
      pa.q[0] = *(const uint4*)(pw2 + half * 8);
      pa.q[1] = *(const uint4*)(pw2 + 16 + half * 8);
    }
    // O += P(16x32) @ V(32x128): 8 WMMAs, B fragments from transposed LDS
#pragma unroll
    for (int sub = 0; sub < 8; ++sub) {
      Frag f;
      const unsigned short* vt = Vt + (sub * 16 + l16) * 32;
      f.q[0] = *(const uint4*)(vt + half * 8);
      f.q[1] = *(const uint4*)(vt + 16 + half * 8);
      acc_o[sub] = wmma_bf16(pa.v, f.v, acc_o[sub]);
    }
  }

  // epilogue: normalize and store bf16
#pragma unroll
  for (int sub = 0; sub < 8; ++sub) {
#pragma unroll
    for (int e = 0; e < 8; ++e) {
      const int qrow = qbase + e + 8 * half;
      const int d = sub * 16 + l16;
      Ob[(size_t)qrow * NQ + h * HD + d] = f32_to_bf16(acc_o[sub][e] / rl[e]);
    }
  }
}

// ---------------- host side ----------------
extern "C" void kernel_launch(void* const* d_in, const int* in_sizes, int n_in,
                              void* d_out, int out_size, void* d_ws, size_t ws_size,
                              hipStream_t stream) {
  (void)in_sizes; (void)n_in; (void)out_size; (void)ws_size;
  const float* hs   = (const float*)d_in[0];
  const float* cosT = (const float*)d_in[1];
  const float* sinT = (const float*)d_in[2];
  // d_in[3] = attention_mask (causal mask applied analytically)
  const float* Wq   = (const float*)d_in[4];
  const float* Wk   = (const float*)d_in[5];
  const float* Wv   = (const float*)d_in[6];
  const float* Wo   = (const float*)d_in[7];
  const float* qnw  = (const float*)d_in[8];
  const float* knw  = (const float*)d_in[9];
  float* out = (float*)d_out;

  char* ws = (char*)d_ws;
  size_t off = 0;
  auto alloc = [&](size_t bytes) -> void* {
    void* p = ws + off;
    off += (bytes + 255) & ~(size_t)255;
    return p;
  };
  unsigned short* Xb  = (unsigned short*)alloc((size_t)S_LEN * HID * 2);
  unsigned short* Wqb = (unsigned short*)alloc((size_t)HID * NQ * 2);
  unsigned short* Wkb = (unsigned short*)alloc((size_t)HID * NKVD * 2);
  unsigned short* Wvb = (unsigned short*)alloc((size_t)HID * NKVD * 2);
  unsigned short* Wob = (unsigned short*)alloc((size_t)NQ * HID * 2);
  float*          Qf  = (float*)alloc((size_t)S_LEN * NQ * 4);
  float*          Kf  = (float*)alloc((size_t)S_LEN * NKVD * 4);
  unsigned short* Qbf = (unsigned short*)alloc((size_t)S_LEN * NQ * 2);
  unsigned short* Kbf = (unsigned short*)alloc((size_t)S_LEN * NKVD * 2);
  unsigned short* Vbf = (unsigned short*)alloc((size_t)S_LEN * NKVD * 2);
  unsigned short* Ab  = (unsigned short*)alloc((size_t)S_LEN * NQ * 2);

  auto cvt = [&](const float* src, unsigned short* dst, size_t n) {
    int n4 = (int)(n / 4);
    cvt_bf16_kernel<<<dim3((n4 + 255) / 256), 256, 0, stream>>>(src, dst, n4);
  };
  cvt(hs, Xb, (size_t)S_LEN * HID);
  cvt(Wq, Wqb, (size_t)HID * NQ);
  cvt(Wk, Wkb, (size_t)HID * NKVD);
  cvt(Wv, Wvb, (size_t)HID * NKVD);
  cvt(Wo, Wob, (size_t)NQ * HID);

  // projections
  gemm_bf16_kernel<false><<<dim3(NQ / 128, S_LEN / 128), 256, 0, stream>>>(
      Xb, Wqb, Qf, S_LEN, NQ, HID);
  gemm_bf16_kernel<false><<<dim3(NKVD / 128, S_LEN / 128), 256, 0, stream>>>(
      Xb, Wkb, Kf, S_LEN, NKVD, HID);
  gemm_bf16_kernel<true><<<dim3(NKVD / 128, S_LEN / 128), 256, 0, stream>>>(
      Xb, Wvb, Vbf, S_LEN, NKVD, HID);

  // RMSNorm + RoPE
  normrope_kernel<<<dim3(S_LEN * NHH / 8), 256, 0, stream>>>(Qf, qnw, cosT, sinT, Qbf, NHH);
  normrope_kernel<<<dim3(S_LEN * NKVH / 8), 256, 0, stream>>>(Kf, knw, cosT, sinT, Kbf, NKVH);

  // attention
  attn_kernel<<<dim3(NHH, S_LEN / 64), 128, 0, stream>>>(Qbf, Kbf, Vbf, Ab);

  // output projection (fp32 out)
  gemm_bf16_kernel<false><<<dim3(HID / 128, S_LEN / 128), 256, 0, stream>>>(
      Ab, Wob, out, S_LEN, HID, NQ);
}
